// MentionAttention_88132728914542
// MI455X (gfx1250) — matrix operand order repaired and checked by reading, not compiled, not run on hardware
//
#include <hip/hip_runtime.h>

typedef __attribute__((ext_vector_type(16))) _Float16 v16h;
typedef __attribute__((ext_vector_type(8)))  _Float16 v8h;
typedef __attribute__((ext_vector_type(4)))  _Float16 v4h;
typedef __attribute__((ext_vector_type(8)))  float    v8f;
typedef __attribute__((ext_vector_type(4)))  unsigned int tdm_g0_t;
typedef __attribute__((ext_vector_type(8)))  int tdm_g1_t;
typedef __attribute__((ext_vector_type(4)))  int tdm_g2_t;

// D = A(16x32 f16) * B(32x16 f16) + C(16x16 f32)
__device__ __forceinline__ v8f wmma_f16(v16h a, v16h b, v8f c) {
  return __builtin_amdgcn_wmma_f32_16x16x32_f16(false, a, false, b, (short)0, c, false, false);
}

// A fragment: lane holds row (lane%16); K values: [kb..kb+7] and [16+kb..16+kb+7], kb=(lane/16)*8
__device__ __forceinline__ v16h load_a_frag(const _Float16* row, int lane) {
  const _Float16* p = row + ((lane >> 4) << 3);
  v8h lo = *(const v8h*)(p);
  v8h hi = *(const v8h*)(p + 16);
  return __builtin_shufflevector(lo, hi, 0,1,2,3,4,5,6,7,8,9,10,11,12,13,14,15);
}

// B fragment: lane holds col (lane%16); K values: [kb..kb+7] and [16+kb..16+kb+7], kb=(lane/16)*8
// `row` points at the (col)-th row of the K-contiguous (transposed) source + kstep base.
__device__ __forceinline__ v16h load_b_frag(const _Float16* row, int lane) {
  const _Float16* p = row + ((lane >> 4) << 3);
  v8h lo = *(const v8h*)(p);
  v8h hi = *(const v8h*)(p + 16);
  return __builtin_shufflevector(lo, hi, 0,1,2,3,4,5,6,7,8,9,10,11,12,13,14,15);
}

// ---------------- Transpose+convert: dst[n][k] = f16(src[k][n]) ----------------
__global__ void __launch_bounds__(256) transpose_f16_kernel(const float* __restrict__ src,
                                                            _Float16* __restrict__ dst,
                                                            int K, int N) {
  int e = blockIdx.x * 256 + threadIdx.x;  // grid covers K*N
  int k = e / N, n = e % N;
  dst[(size_t)n * K + k] = (_Float16)src[e];
}

// ---------------- Kernel: qp = f16(relu(querys) @ Wq) [2048 x 256], K=768 ----------------
__device__ __forceinline__ v16h build_a_f32_relu(const float* row, int lane) {
  const float* p = row + ((lane >> 4) << 3);
  v16h r;
#pragma unroll
  for (int j = 0; j < 8; ++j) { float x = p[j];      r[j]     = (_Float16)(x > 0.f ? x : 0.f); }
#pragma unroll
  for (int j = 0; j < 8; ++j) { float x = p[16 + j]; r[8 + j] = (_Float16)(x > 0.f ? x : 0.f); }
  return r;
}

__global__ void __launch_bounds__(256) qp_proj_kernel(const float* __restrict__ querys,
                                                      const _Float16* __restrict__ WqT,
                                                      _Float16* __restrict__ qp) {
  const int tid = threadIdx.x, lane = tid & 31, wave = tid >> 5;
  const int lm = lane & 15, lh = lane >> 4;
  const int tile = blockIdx.x * 8 + wave;      // 2048 tiles of 16x16
  const int q0 = (tile >> 4) << 4;             // row block over B*E = 2048
  const int n0 = (tile & 15) << 4;             // col block over DK = 256
  v8f acc = {};
  for (int ks = 0; ks < 24; ++ks) {            // K = 768
    v16h a  = build_a_f32_relu(querys + (size_t)(q0 + lm) * 768 + ks * 32, lane);
    v16h bf = load_b_frag(WqT + (size_t)(n0 + lm) * 768 + ks * 32, lane);
    acc = wmma_f16(a, bf, acc);
  }
#pragma unroll
  for (int r = 0; r < 8; ++r)
    qp[(size_t)(q0 + r + (lh << 3)) * 256 + n0 + lm] = (_Float16)acc[r];
}

// ---------------- Fused per-(b,a) attention ----------------
// LDS layout (bytes):
//  keysRM (relu, f16, [64][256])   @ 0       (32768)
//  keysT  (raw,  f16, [256][64])   @ 32768   (32768)
//  stage  (raw keys f32, 64KB)     @ 65536   (65536)  [dead after conversion; aliased by:]
//    kp   (f16,  [64][256])        @ 65536   (32768)
//    attn (f16,  [128][64])        @ 98304   (16384)
//  wm     (f32,  [64])             @ 131072  (256)
//  mask   (i32,  [64])             @ 131328  (256)
#define SMEM_BYTES 131584

__global__ void __launch_bounds__(256, 1) mention_attn_kernel(
    const float* __restrict__ keys, const _Float16* __restrict__ qp,
    const _Float16* __restrict__ WkT, const float* __restrict__ Wm_w,
    const float* __restrict__ Wm_b, const int* __restrict__ masks,
    float* __restrict__ out) {
  extern __shared__ char smem[];
  _Float16* keysRM = (_Float16*)(smem);
  _Float16* keysT  = (_Float16*)(smem + 32768);
  float*    stage  = (float*)(smem + 65536);
  _Float16* kpS    = (_Float16*)(smem + 65536);
  _Float16* attnS  = (_Float16*)(smem + 98304);
  float*    wmS    = (float*)(smem + 131072);
  int*      maskS  = (int*)(smem + 131328);

  const int tid = threadIdx.x, lane = tid & 31, wave = tid >> 5;
  const int lm = lane & 15, lh = lane >> 4;
  const int ba = blockIdx.x;                 // b*128 + a
  const int b  = ba >> 7;
  const float* keysG = keys + (size_t)ba * (64 * 256);

  // Warm L2/L0 for this wave's phase-2 A operand (qp rows).
  __builtin_prefetch((const void*)(qp + (size_t)(b * 128 + wave * 16 + lm) * 256 + lh * 128), 0, 0);

  // ---- Phase 0: TDM DMA keys[b,a] (16384 f32 = 64KB) global -> LDS staging ----
  if (wave == 0) {
    unsigned long long ga = (unsigned long long)(uintptr_t)keysG;
    tdm_g0_t g0;
    g0[0] = 1u;                                       // count=1, is_restore=0, gather=0
    g0[1] = (unsigned)(uintptr_t)stage;               // lds_addr (flat low 32 = LDS byte addr)
    g0[2] = (unsigned)(ga & 0xFFFFFFFFu);             // global_addr[31:0]
    g0[3] = (unsigned)((ga >> 32) & 0x1FFFFFFu)       // global_addr[56:32]
            | (2u << 30);                             // type=2 (image)
    tdm_g1_t g1;
    g1[0] = 0x20000;     // workgroup_mask=0, data_size=2 (4 bytes)
    g1[1] = 0x40000000;  // tensor_dim0[15:0]=16384 in bits 63:48
    g1[2] = 0x10000;     // tensor_dim0[31:16]=0, tensor_dim1=1
    g1[3] = 0x40000000;  // tile_dim0=16384 in bits 127:112
    g1[4] = 1;           // tile_dim1=1, tile_dim2=0
    g1[5] = 16384;       // tensor_dim0_stride[31:0]
    g1[6] = 0;           // tensor_dim0_stride[47:32]=0, tensor_dim1_stride lo=0
    g1[7] = 0;
    tdm_g2_t gz4 = {0, 0, 0, 0};
    tdm_g1_t gz8 = {0, 0, 0, 0, 0, 0, 0, 0};
    __builtin_amdgcn_tensor_load_to_lds(g0, g1, gz4, gz4, gz8, 0);
    __builtin_amdgcn_s_wait_tensorcnt((short)0);
  }
  if (tid < 64) maskS[tid] = masks[(size_t)ba * 64 + tid];
  __syncthreads();

  // ---- Phase 0b: convert staged f32 keys -> relu'd RM f16 + raw transposed f16 ----
#pragma unroll 4
  for (int i = 0; i < 16; ++i) {
    int off = i * 1024 + tid * 4;
    const float4 v = *(const float4*)(stage + off);
    int m = off >> 8, d = off & 255;
    v4h rel;
    rel[0] = (_Float16)(v.x > 0.f ? v.x : 0.f);
    rel[1] = (_Float16)(v.y > 0.f ? v.y : 0.f);
    rel[2] = (_Float16)(v.z > 0.f ? v.z : 0.f);
    rel[3] = (_Float16)(v.w > 0.f ? v.w : 0.f);
    *(v4h*)(keysRM + m * 256 + d) = rel;
    keysT[(d + 0) * 64 + m] = (_Float16)v.x;
    keysT[(d + 1) * 64 + m] = (_Float16)v.y;
    keysT[(d + 2) * 64 + m] = (_Float16)v.z;
    keysT[(d + 3) * 64 + m] = (_Float16)v.w;
  }
  __syncthreads();

  // ---- Phase 1a: wm[m] = relu(keys[m,:]) . Wm_w + b   (4 threads per m) ----
  {
    int m = tid >> 2, p0 = (tid & 3) * 64;
    const _Float16* kr = keysRM + m * 256 + p0;
    float s = 0.f;
    for (int k = 0; k < 64; ++k) s += (float)kr[k] * Wm_w[p0 + k];
    s += __shfl_xor(s, 1, 32);
    s += __shfl_xor(s, 2, 32);
    if ((tid & 3) == 0) wmS[m] = s + Wm_b[0];
  }
  // ---- Phase 1b: kp = relu(keys) @ Wk -> LDS f16 [64][256]; 8 tiles per wave ----
  // (kpS aliases the now-dead f32 staging buffer; safe after the barrier above)
  {
    const int mt = wave & 3, nh = wave >> 2;   // rows mt*16.., cols nh*128..
    v8f acc[8] = {};
    for (int ks = 0; ks < 8; ++ks) {           // K = 256
      v16h a = load_a_frag(keysRM + (mt * 16 + lm) * 256 + ks * 32, lane);
#pragma unroll
      for (int nt = 0; nt < 8; ++nt) {
        v16h bf = load_b_frag(WkT + (size_t)(nh * 128 + nt * 16 + lm) * 256 + ks * 32, lane);
        acc[nt] = wmma_f16(a, bf, acc[nt]);
      }
    }
#pragma unroll
    for (int nt = 0; nt < 8; ++nt)
#pragma unroll
      for (int r = 0; r < 8; ++r)
        kpS[(mt * 16 + r + (lh << 3)) * 256 + nh * 128 + nt * 16 + lm] = (_Float16)acc[nt][r];
  }
  __syncthreads();

  // ---- Phase 2: scores = qp[b] @ kp^T, masked softmax over M=64; one q-tile per wave ----
  {
    const int q0 = wave * 16;
    v8f acc[4] = {};
    for (int ks = 0; ks < 8; ++ks) {           // K = 256 (d)
      v16h a = load_a_frag(qp + (size_t)(b * 128 + q0 + lm) * 256 + ks * 32, lane);
#pragma unroll
      for (int c = 0; c < 4; ++c) {
        v16h bf = load_b_frag(kpS + (c * 16 + lm) * 256 + ks * 32, lane);
        acc[c] = wmma_f16(a, bf, acc[c]);
      }
    }
    const float scale = 0.03608439182435161f;  // 1/sqrt(768)
    float bias[4]; int mk[4];
#pragma unroll
    for (int c = 0; c < 4; ++c) { bias[c] = wmS[c * 16 + lm]; mk[c] = maskS[c * 16 + lm]; }
#pragma unroll
    for (int r = 0; r < 8; ++r) {
      float x[4];
#pragma unroll
      for (int c = 0; c < 4; ++c)
        x[c] = mk[c] ? (acc[c][r] * scale + bias[c]) : -1e12f;
      float mx = fmaxf(fmaxf(x[0], x[1]), fmaxf(x[2], x[3]));
      mx = fmaxf(mx, __shfl_xor(mx, 1, 32));
      mx = fmaxf(mx, __shfl_xor(mx, 2, 32));
      mx = fmaxf(mx, __shfl_xor(mx, 4, 32));
      mx = fmaxf(mx, __shfl_xor(mx, 8, 32));   // row's 64 cols live in one 16-lane half
      float sum = 0.f;
#pragma unroll
      for (int c = 0; c < 4; ++c) { x[c] = __expf(x[c] - mx); sum += x[c]; }
      sum += __shfl_xor(sum, 1, 32);
      sum += __shfl_xor(sum, 2, 32);
      sum += __shfl_xor(sum, 4, 32);
      sum += __shfl_xor(sum, 8, 32);
      float inv = 1.f / sum;
#pragma unroll
      for (int c = 0; c < 4; ++c) acc[c][r] = x[c] * inv;
    }
#pragma unroll
    for (int c = 0; c < 4; ++c)
#pragma unroll
      for (int r = 0; r < 8; ++r)
        attnS[(q0 + r + (lh << 3)) * 64 + c * 16 + lm] = (_Float16)acc[c][r];
  }
  __syncthreads();

  // ---- Phase 3: out[b,a] = attn(128x64) @ keys(64x256), f32 stores ----
  {
    const int q0 = wave * 16;
    float* outG = out + (size_t)ba * (128 * 256);
    for (int dt = 0; dt < 16; ++dt) {
      v8f acc = {};
#pragma unroll
      for (int ks = 0; ks < 2; ++ks) {         // K = 64 (m)
        v16h a  = load_a_frag(attnS + (q0 + lm) * 64 + ks * 32, lane);
        v16h bf = load_b_frag(keysT + (dt * 16 + lm) * 64 + ks * 32, lane);
        acc = wmma_f16(a, bf, acc);
      }
#pragma unroll
      for (int r = 0; r < 8; ++r)
        outG[(size_t)(q0 + r + (lh << 3)) * 256 + dt * 16 + lm] = acc[r];
    }
  }
}

extern "C" void kernel_launch(void* const* d_in, const int* in_sizes, int n_in,
                              void* d_out, int out_size, void* d_ws, size_t ws_size,
                              hipStream_t stream) {
  const float* querys = (const float*)d_in[0];
  const float* keys   = (const float*)d_in[1];
  const float* Wq     = (const float*)d_in[2];
  const float* Wk     = (const float*)d_in[3];
  const float* Wm_w   = (const float*)d_in[4];
  const float* Wm_b   = (const float*)d_in[5];
  const int*   masks  = (const int*)d_in[6];
  float* out = (float*)d_out;

  char* ws = (char*)d_ws;
  _Float16* qpF = (_Float16*)ws;                               // 2048*256 f16 = 1 MB
  _Float16* WkT = (_Float16*)(ws + (size_t)1048576);           // 256*256 f16  = 128 KB
  _Float16* WqT = (_Float16*)(ws + (size_t)1048576 + 131072);  // 256*768 f16  = 384 KB

  transpose_f16_kernel<<<768, 256, 0, stream>>>(Wq, WqT, 768, 256);
  transpose_f16_kernel<<<256, 256, 0, stream>>>(Wk, WkT, 256, 256);
  qp_proj_kernel<<<256, 256, 0, stream>>>(querys, WqT, qpF);
  mention_attn_kernel<<<2048, 256, SMEM_BYTES, stream>>>(keys, qpF, WkT, Wm_w, Wm_b,
                                                         masks, out);
}
